// LIGN_CIFAR_23210003268065
// MI455X (gfx1250) — matrix operand order repaired and checked by reading, not compiled, not run on hardware
//
#include <hip/hip_runtime.h>
#include <cmath>

// GCN pipeline for MI455X (gfx1250, wave32).
// Layers 1-2: split-f16 GEMM (x = hi + lo, ~22-bit mantissa) on
//   v_wmma_f32_16x16x32_f16, 2x2 output tiles per wave (32x32) so each A/B
//   fragment feeds 2 WMMAs: 16 b128 loads per 12 wmma per K=32 chunk.
// Layers 3-4: exact f32 WMMA 16x16x4.
// All weights/biases pre-padded -> branchless inner loops (EXEC all-1 at WMMA).

typedef __attribute__((ext_vector_type(16))) _Float16 v16h;
typedef __attribute__((ext_vector_type(8)))  _Float16 v8h;
typedef __attribute__((ext_vector_type(8)))  float    v8f;
typedef __attribute__((ext_vector_type(2)))  float    v2f;

#define NNODES 20000
#define NEDGES 320000

#define CAT16(a, b) __builtin_shufflevector(a, b, 0,1,2,3,4,5,6,7,8,9,10,11,12,13,14,15)
#define WMMA16(A, B, C) __builtin_amdgcn_wmma_f32_16x16x32_f16(false, A, false, B, (short)0, C, false, false)

// ---------------- split f32 -> f16 hi/lo (activations) ----------------
template<bool RELU>
__global__ __launch_bounds__(256) void split_f16_kernel(
    const float* __restrict__ in, int ldin, int Kin,
    _Float16* __restrict__ hi, _Float16* __restrict__ lo, int ldout)
{
    int r = blockIdx.x;
    int k = blockIdx.y * blockDim.x + threadIdx.x;
    if (k >= ldout) return;
    float x = 0.0f;
    if (k < Kin) {
        x = in[(size_t)r * ldin + k];
        if (RELU) x = fmaxf(x, 0.0f);
    }
    _Float16 h = (_Float16)x;
    _Float16 l = (_Float16)(x - (float)h);
    size_t o = (size_t)r * ldout + k;
    hi[o] = h;
    lo[o] = l;
}

// ---- weights: row-major f32 [K x N] -> col-major f16 hi/lo [Ncols x Kpad] ----
__global__ __launch_bounds__(256) void wsplit_f16_kernel(
    const float* __restrict__ W, int K, int N,
    _Float16* __restrict__ hi, _Float16* __restrict__ lo, int Kpad)
{
    int col = blockIdx.x;
    int k = blockIdx.y * blockDim.x + threadIdx.x;
    if (k >= Kpad) return;
    float x = (col < N && k < K) ? W[(size_t)k * N + col] : 0.0f;
    _Float16 h = (_Float16)x;
    _Float16 l = (_Float16)(x - (float)h);
    size_t o = (size_t)col * Kpad + k;
    hi[o] = h;
    lo[o] = l;
}

// ---- weights for f32 path: [K x N] -> zero-col-padded [K x ldw] ----
__global__ __launch_bounds__(256) void padw_f32_kernel(
    const float* __restrict__ W, int K, int N, float* __restrict__ Wp, int ldw)
{
    int i = blockIdx.x * blockDim.x + threadIdx.x;
    if (i >= K * ldw) return;
    int k = i / ldw, n = i - k * ldw;
    Wp[i] = (n < N) ? W[(size_t)k * N + n] : 0.0f;
}

__global__ void padb_kernel(const float* __restrict__ b, int N,
                            float* __restrict__ bp, int ldc)
{
    int i = blockIdx.x * blockDim.x + threadIdx.x;
    if (i < ldc) bp[i] = (i < N) ? b[i] : 0.0f;
}

// ------- split-f16 GEMM, 2x2 tiles (32x32 C) per wave; K, tiles even -------
__global__ __launch_bounds__(256) void gemm_wmma_f16split_2x2(
    const _Float16* __restrict__ Ahi, const _Float16* __restrict__ Alo, int lda,
    const _Float16* __restrict__ Bhi, const _Float16* __restrict__ Blo, int ldb,
    const float* __restrict__ biasP, float* __restrict__ C, int ldc,
    int K, int tilesN2, int numWaves)
{
    int wave = blockIdx.x * (blockDim.x >> 5) + ((int)threadIdx.x >> 5);
    if (wave >= numWaves) return;                  // wave-uniform exit
    int tm = wave / tilesN2;
    int tn = wave - tm * tilesN2;
    int m0 = tm << 5, n0 = tn << 5;
    int lane = threadIdx.x & 31;
    int l16  = lane & 15;
    int hh   = lane >> 4;
    int col0 = n0 + l16, col1 = col0 + 16;

    v8f c00, c01, c10, c11;
    float bv0 = biasP[col0], bv1 = biasP[col1];
#pragma unroll
    for (int i = 0; i < 8; ++i) { c00[i] = bv0; c01[i] = bv1; c10[i] = bv0; c11[i] = bv1; }

    // A fragment: lo-lanes K {k..k+7, k+16..k+23}, hi-lanes +8
    const _Float16* pAh0 = Ahi + (size_t)(m0 + l16) * lda + (hh << 3);
    const _Float16* pAl0 = Alo + (size_t)(m0 + l16) * lda + (hh << 3);
    const _Float16* pAh1 = pAh0 + (size_t)16 * lda;
    const _Float16* pAl1 = pAl0 + (size_t)16 * lda;
    // B fragment: lo-lanes K {k..k+7, k+16..k+23}, hi-lanes +8; col-major B
    const _Float16* pBh0 = Bhi + (size_t)col0 * ldb + (hh << 3);
    const _Float16* pBl0 = Blo + (size_t)col0 * ldb + (hh << 3);
    const _Float16* pBh1 = Bhi + (size_t)col1 * ldb + (hh << 3);
    const _Float16* pBl1 = Blo + (size_t)col1 * ldb + (hh << 3);

    for (int k = 0; k < K; k += 32) {
        v16h ah0 = CAT16(*(const v8h*)(pAh0 + k), *(const v8h*)(pAh0 + k + 16));
        v16h ah1 = CAT16(*(const v8h*)(pAh1 + k), *(const v8h*)(pAh1 + k + 16));
        v16h al0 = CAT16(*(const v8h*)(pAl0 + k), *(const v8h*)(pAl0 + k + 16));
        v16h al1 = CAT16(*(const v8h*)(pAl1 + k), *(const v8h*)(pAl1 + k + 16));
        v16h bh0 = CAT16(*(const v8h*)(pBh0 + k), *(const v8h*)(pBh0 + k + 16));
        v16h bh1 = CAT16(*(const v8h*)(pBh1 + k), *(const v8h*)(pBh1 + k + 16));
        v16h bl0 = CAT16(*(const v8h*)(pBl0 + k), *(const v8h*)(pBl0 + k + 16));
        v16h bl1 = CAT16(*(const v8h*)(pBl1 + k), *(const v8h*)(pBl1 + k + 16));
        // hi*hi
        c00 = WMMA16(ah0, bh0, c00);
        c01 = WMMA16(ah0, bh1, c01);
        c10 = WMMA16(ah1, bh0, c10);
        c11 = WMMA16(ah1, bh1, c11);
        // hi*lo
        c00 = WMMA16(ah0, bl0, c00);
        c01 = WMMA16(ah0, bl1, c01);
        c10 = WMMA16(ah1, bl0, c10);
        c11 = WMMA16(ah1, bl1, c11);
        // lo*hi
        c00 = WMMA16(al0, bh0, c00);
        c01 = WMMA16(al0, bh1, c01);
        c10 = WMMA16(al1, bh0, c10);
        c11 = WMMA16(al1, bh1, c11);
    }

#pragma unroll
    for (int i = 0; i < 8; ++i) {
        size_t r0 = (size_t)(m0 + (hh << 3) + i) * ldc;
        size_t r1 = (size_t)(m0 + 16 + (hh << 3) + i) * ldc;
        C[r0 + col0] = c00[i];
        C[r0 + col1] = c01[i];
        C[r1 + col0] = c10[i];
        C[r1 + col1] = c11[i];
    }
}

// ---------------- exact f32 GEMM (layers 3-4), padded weights ----------------
template<bool RELU>
__global__ __launch_bounds__(256) void gemm_wmma_f32(
    const float* __restrict__ A, int lda,
    const float* __restrict__ Wp, int ldw, const float* __restrict__ biasP,
    float* __restrict__ C, int ldc,
    int K, int tilesN, int numTiles)
{
    int wave = blockIdx.x * (blockDim.x >> 5) + ((int)threadIdx.x >> 5);
    if (wave >= numTiles) return;
    int tm = wave / tilesN;
    int tn = wave - tm * tilesN;
    int m0 = tm << 4, n0 = tn << 4;
    int lane = threadIdx.x & 31;
    int l16  = lane & 15;
    int hh   = lane >> 4;
    int khalf = hh << 1;
    int col  = n0 + l16;

    v8f c;
    float bv = biasP[col];
#pragma unroll
    for (int i = 0; i < 8; ++i) c[i] = bv;

    const float* Arow = A + (size_t)(m0 + l16) * lda;
    for (int k = 0; k < K; k += 4) {
        int ka = k + khalf;
        v2f a;
        a.x = Arow[ka];
        a.y = Arow[ka + 1];
        if (RELU) {
            a.x = fmaxf(a.x, 0.0f);
            a.y = fmaxf(a.y, 0.0f);
        }
        v2f b;
        b.x = Wp[(size_t)ka * ldw + col];
        b.y = Wp[(size_t)(ka + 1) * ldw + col];
        c = __builtin_amdgcn_wmma_f32_16x16x4_f32(false, a, false, b, (short)0, c, false, false);
    }
#pragma unroll
    for (int i = 0; i < 8; ++i)
        C[(size_t)(m0 + (hh << 3) + i) * ldc + col] = c[i];
}

// ---------------- aggregation / epilogue ----------------
__global__ __launch_bounds__(256) void copy4_kernel(
    const float4* __restrict__ in, float4* __restrict__ out, int n4)
{
    int i = blockIdx.x * blockDim.x + threadIdx.x;
    if (i < n4) out[i] = in[i];
}

__global__ void edge_scatter_kernel(
    const float* __restrict__ h, float* __restrict__ a,
    const int* __restrict__ src, const int* __restrict__ dst,
    int N, int ldc, int shift)
{
    int e   = blockIdx.x >> shift;
    int blk = blockIdx.x & ((1 << shift) - 1);
    int f   = blk * blockDim.x + threadIdx.x;
    if (f >= N) return;
    int s = src[e];
    int d = dst[e];
    atomicAdd(&a[(size_t)d * ldc + f], h[(size_t)s * ldc + f]);
}

__global__ __launch_bounds__(256) void tanh_out_kernel(
    const float* __restrict__ a, float* __restrict__ out)
{
    int i = blockIdx.x * blockDim.x + threadIdx.x;
    if (i >= NNODES * 10) return;
    int node = i / 10;
    int f = i - node * 10;
    out[i] = tanhf(a[(size_t)node * 16 + f]);
}

static inline size_t align256(size_t x) { return (x + 255) & ~(size_t)255; }

extern "C" void kernel_launch(void* const* d_in, const int* in_sizes, int n_in,
                              void* d_out, int out_size, void* d_ws, size_t ws_size,
                              hipStream_t stream) {
    (void)in_sizes; (void)n_in; (void)out_size; (void)ws_size;

    const float* features = (const float*)d_in[0];
    const int*   src      = (const int*)  d_in[1];
    const int*   dst      = (const int*)  d_in[2];
    const float* W1 = (const float*)d_in[3];  const float* b1 = (const float*)d_in[4];
    const float* W2 = (const float*)d_in[5];  const float* b2 = (const float*)d_in[6];
    const float* W3 = (const float*)d_in[7];  const float* b3 = (const float*)d_in[8];
    const float* W4 = (const float*)d_in[9];  const float* b4 = (const float*)d_in[10];
    float* out = (float*)d_out;

    const int LD1 = 1024, LD2 = 320, LD3 = 32, LD4 = 16;   // padded widths (LD1/16, LD2/16 even)
    const int KP1 = 3072, KP2 = 1024;                      // padded K (mult of 32)

    // ---------------- workspace layout (aliased across layer timeline) ----------
    char* base = (char*)d_ws;
    float* bufA = (float*)base;                                       // h buffer (81.92 MB)
    char* r1 = base + align256((size_t)NNODES * LD1 * 4);
    // Phase 1 use of r1: layer-1 split A (2 x 122.88 MB)
    _Float16* A1hi = (_Float16*)r1;
    _Float16* A1lo = A1hi + (size_t)NNODES * KP1;
    // Phase 2 use of r1 (after GEMM-1): a buffer + layer-2 split A
    float* bufB = (float*)r1;
    _Float16* A2hi = (_Float16*)(r1 + align256((size_t)NNODES * LD1 * 4));
    _Float16* A2lo = A2hi + (size_t)NNODES * KP2;
    // Weights / biases (persistent through the call)
    char* wp = r1 + (size_t)NNODES * KP1 * 2 * 2;                     // +245.76 MB
    _Float16* Wt1hi = (_Float16*)wp;  wp += align256((size_t)LD1 * KP1 * 2);
    _Float16* Wt1lo = (_Float16*)wp;  wp += align256((size_t)LD1 * KP1 * 2);
    _Float16* Wt2hi = (_Float16*)wp;  wp += align256((size_t)LD2 * KP2 * 2);
    _Float16* Wt2lo = (_Float16*)wp;  wp += align256((size_t)LD2 * KP2 * 2);
    float* W3p = (float*)wp;          wp += align256((size_t)300 * LD3 * 4);
    float* W4p = (float*)wp;          wp += align256((size_t)20 * LD4 * 4);
    float* b1p = (float*)wp;          wp += 256 * 16;
    float* b2p = (float*)wp;          wp += 256 * 8;
    float* b3p = (float*)wp;          wp += 256;
    float* b4p = (float*)wp;          wp += 256;

    // ---------------- weight / bias preparation ----------------
    wsplit_f16_kernel<<<dim3(LD1, KP1 / 256), 256, 0, stream>>>(W1, 3072, 1000, Wt1hi, Wt1lo, KP1);
    wsplit_f16_kernel<<<dim3(LD2, KP2 / 256), 256, 0, stream>>>(W2, 1000, 300, Wt2hi, Wt2lo, KP2);
    padw_f32_kernel<<<(300 * LD3 + 255) / 256, 256, 0, stream>>>(W3, 300, 20, W3p, LD3);
    padw_f32_kernel<<<(20 * LD4 + 255) / 256, 256, 0, stream>>>(W4, 20, 10, W4p, LD4);
    padb_kernel<<<(LD1 + 255) / 256, 256, 0, stream>>>(b1, 1000, b1p, LD1);
    padb_kernel<<<(LD2 + 255) / 256, 256, 0, stream>>>(b2, 300, b2p, LD2);
    padb_kernel<<<1, 64, 0, stream>>>(b3, 20, b3p, LD3);
    padb_kernel<<<1, 64, 0, stream>>>(b4, 10, b4p, LD4);

    // ---------------- Layer 1: K=3072, N=1000, split-f16 WMMA 2x2 --------------
    {
        split_f16_kernel<false><<<dim3(NNODES, KP1 / 256), 256, 0, stream>>>(
            features, 3072, 3072, A1hi, A1lo, KP1);
        int tilesN2 = LD1 / 32, numWaves = (NNODES / 32) * tilesN2;   // 32, 20000
        gemm_wmma_f16split_2x2<<<(numWaves + 7) / 8, 256, 0, stream>>>(
            A1hi, A1lo, KP1, Wt1hi, Wt1lo, KP1, b1p, bufA, LD1, KP1, tilesN2, numWaves);
        int n4 = NNODES * LD1 / 4;
        copy4_kernel<<<(n4 + 255) / 256, 256, 0, stream>>>((const float4*)bufA, (float4*)bufB, n4);
        edge_scatter_kernel<<<NEDGES << 2, 256, 0, stream>>>(bufA, bufB, src, dst, 1000, LD1, 2);
    }
    // ---------------- Layer 2: K=1000->1024, N=300, relu fused into split ------
    {
        split_f16_kernel<true><<<dim3(NNODES, KP2 / 256), 256, 0, stream>>>(
            bufB, LD1, 1000, A2hi, A2lo, KP2);
        int tilesN2 = LD2 / 32, numWaves = (NNODES / 32) * tilesN2;   // 10, 6250
        gemm_wmma_f16split_2x2<<<(numWaves + 7) / 8, 256, 0, stream>>>(
            A2hi, A2lo, KP2, Wt2hi, Wt2lo, KP2, b2p, bufA, LD2, KP2, tilesN2, numWaves);
        int n4 = NNODES * LD2 / 4;
        copy4_kernel<<<(n4 + 255) / 256, 256, 0, stream>>>((const float4*)bufA, (float4*)bufB, n4);
        edge_scatter_kernel<<<NEDGES << 1, 256, 0, stream>>>(bufA, bufB, src, dst, 300, LD2, 1);
    }
    // ---------------- Layer 3: K=300, N=20, exact f32 WMMA, relu on input ------
    {
        int tilesN = LD3 / 16, numTiles = (NNODES / 16) * tilesN;     // 2, 2500
        gemm_wmma_f32<true><<<(numTiles + 7) / 8, 256, 0, stream>>>(
            bufB, LD2, W3p, LD3, b3p, bufA, LD3, 300, tilesN, numTiles);
        int n4 = NNODES * LD3 / 4;
        copy4_kernel<<<(n4 + 255) / 256, 256, 0, stream>>>((const float4*)bufA, (float4*)bufB, n4);
        edge_scatter_kernel<<<NEDGES, 32, 0, stream>>>(bufA, bufB, src, dst, 20, LD3, 0);
    }
    // ---------------- Layer 4: K=20, N=10, exact f32 WMMA, no relu -------------
    {
        int tilesN = LD4 / 16, numTiles = (NNODES / 16) * tilesN;     // 1, 1250
        gemm_wmma_f32<false><<<(numTiles + 7) / 8, 256, 0, stream>>>(
            bufB, LD3, W4p, LD4, b4p, bufA, LD4, 20, tilesN, numTiles);
        int n4 = NNODES * LD4 / 4;
        copy4_kernel<<<(n4 + 255) / 256, 256, 0, stream>>>((const float4*)bufA, (float4*)bufB, n4);
        edge_scatter_kernel<<<NEDGES, 32, 0, stream>>>(bufA, bufB, src, dst, 10, LD4, 0);
    }
    // ---------------- tanh epilogue ----------------
    tanh_out_kernel<<<(NNODES * 10 + 255) / 256, 256, 0, stream>>>(bufB, out);
}
